// AtomicCharge_77781857730661
// MI455X (gfx1250) — compile-verified, matrix operand order checked by reading, not run on hardware
//
#include <hip/hip_runtime.h>

typedef __attribute__((ext_vector_type(16))) _Float16 v16h;
typedef __attribute__((ext_vector_type(8)))  _Float16 v8h;
typedef __attribute__((ext_vector_type(4)))  _Float16 v4h;
typedef __attribute__((ext_vector_type(8)))  float    v8f;
typedef __attribute__((ext_vector_type(4)))  float    v4f;
typedef __attribute__((ext_vector_type(2)))  float    v2f;

#define NODE_DIM 128
#define HID      64
#define MT       256      // atoms per block
#define XPAD     136      // f16 elems per padded row (272B, 16B-aligned, conflict-padded)
#define HPAD     66       // f32 elems per padded h row (264B, 8B-aligned, conflict-free b64)

// ---------------------------------------------------------------------------
// Kernel 1: per-atom MLP  out[a] = (silu(x@W1 + b1) @ W2 + b2)
// f16 WMMA with f32 accumulate; x/W1 converted on the fly.
// ---------------------------------------------------------------------------
__global__ void __launch_bounds__(256)
k_mlp(const float* __restrict__ x, const float* __restrict__ W1,
      const float* __restrict__ b1, const float* __restrict__ W2,
      const float* __restrict__ b2, float* __restrict__ out, int n)
{
    // 256*68 dwords = 69632 B : holds x-tile as f16, later reused for h as f32
    __shared__ float smemX[MT * XPAD / 2];
    __shared__ float smemW[HID * XPAD / 2];       // W1^T as f16, 17408 B
    __shared__ float sB1[HID];
    __shared__ float sW2[HID];
    __shared__ float sB2;

    _Float16* sX  = (_Float16*)smemX;
    float*    sH  = smemX;                        // aliased after barrier (256*66 <= 256*68)
    _Float16* sWT = (_Float16*)smemW;

    const int tid = threadIdx.x;
    const int a0  = blockIdx.x * MT;

    // ---- stage W1^T (f16): coalesced global read, transposed LDS write ---
    for (int i = tid; i < NODE_DIM * HID; i += 256) {
        int k  = i >> 6;            // W1 row   0..127
        int nn = i & 63;            // W1 col   0..63
        sWT[nn * XPAD + k] = (_Float16)W1[i];
    }
    if (tid < HID) { sB1[tid] = b1[tid]; sW2[tid] = W2[tid]; }
    if (tid == 0)  { sB2 = b2[0]; }

    // ---- stage x tile (fp32 -> f16), zero-fill OOB rows ------------------
    for (int i = tid; i < MT * (NODE_DIM / 4); i += 256) {
        int row = i >> 5;           // 0..255
        int c4  = i & 31;           // float4 column
        int ga  = a0 + row;
        v4f v = {0.f, 0.f, 0.f, 0.f};
        if (ga < n) v = *(const v4f*)(x + (size_t)ga * NODE_DIM + (size_t)c4 * 4);
        v4h h4;
        h4.x = (_Float16)v.x; h4.y = (_Float16)v.y;
        h4.z = (_Float16)v.z; h4.w = (_Float16)v.w;
        *(v4h*)(sX + row * XPAD + c4 * 4) = h4;    // 8B-aligned (272 % 8 == 0)
    }
    __syncthreads();

    // ---- WMMA: wave w owns M-tiles {2w, 2w+1}, all 4 N-tiles -------------
    const int lane = tid & 31;
    const int wv   = tid >> 5;      // 0..7
    const int hf   = lane >> 4;     // half-wave: 0/1
    const int idx  = lane & 15;

    v8f acc[2][4] = {};

    for (int kc = 0; kc < 4; ++kc) {            // K chunks of 32
        v16h afr[2];
#pragma unroll
        for (int t = 0; t < 2; ++t) {
            int row = (2 * wv + t) * 16 + idx;  // A: lane idx -> row M
            const _Float16* p = sX + row * XPAD + kc * 32 + hf * 8;
            v8h lo = *(const v8h*)p;            // K 0..7 (hf=0) / 8..15 (hf=1)
            v8h hi = *(const v8h*)(p + 16);     // K 16..23 / 24..31
            v16h a;
#pragma unroll
            for (int e = 0; e < 8; ++e) { a[e] = lo[e]; a[8 + e] = hi[e]; }
            afr[t] = a;
        }
#pragma unroll
        for (int nt = 0; nt < 4; ++nt) {
            int nn = nt * 16 + idx;             // B: lane idx -> column N
            const _Float16* p = sWT + nn * XPAD + kc * 32 + hf * 8;
            v8h lo = *(const v8h*)p;
            v8h hi = *(const v8h*)(p + 16);
            v16h bfr;
#pragma unroll
            for (int e = 0; e < 8; ++e) { bfr[e] = lo[e]; bfr[8 + e] = hi[e]; }
            acc[0][nt] = __builtin_amdgcn_wmma_f32_16x16x32_f16(
                false, afr[0], false, bfr, (short)0, acc[0][nt], false, false);
            acc[1][nt] = __builtin_amdgcn_wmma_f32_16x16x32_f16(
                false, afr[1], false, bfr, (short)0, acc[1][nt], false, false);
        }
    }
    __syncthreads();   // everyone done reading sX before h overwrites it

    // ---- spill C tiles to LDS h[256][64]: lane L VGPR r -> (M=r+8*hf, N=idx)
#pragma unroll
    for (int t = 0; t < 2; ++t)
#pragma unroll
        for (int nt = 0; nt < 4; ++nt)
#pragma unroll
            for (int r = 0; r < 8; ++r) {
                int m  = (2 * wv + t) * 16 + hf * 8 + r;
                int nn = nt * 16 + idx;
                sH[m * HPAD + nn] = acc[t][nt][r];
            }
    __syncthreads();

    // ---- silu + dot(W2): one thread per atom, b64 LDS reads --------------
    {
        int ga = a0 + tid;
        if (ga < n) {
            const float* hrow = sH + tid * HPAD;   // 264B row stride: 8B-aligned
            float s = sB2;
#pragma unroll
            for (int j = 0; j < HID; j += 2) {
                v2f hv2 = *(const v2f*)(hrow + j);
                v2f b2v = *(const v2f*)(sB1 + j);
                v2f w2v = *(const v2f*)(sW2 + j);
                float h0 = hv2.x + b2v.x;
                float h1 = hv2.y + b2v.y;
                s += h0 * (1.0f / (1.0f + __expf(-h0))) * w2v.x;
                s += h1 * (1.0f / (1.0f + __expf(-h1))) * w2v.y;
            }
            out[ga] = s;
        }
    }
}

// ---------------------------------------------------------------------------
// Kernel 2: deterministic per-molecule segment sums (batch is sorted)
// ---------------------------------------------------------------------------
__global__ void __launch_bounds__(256)
k_molsum(const float* __restrict__ atom_out,
         const int* __restrict__ batch,
         int n_atoms, int n_mol,
         float* __restrict__ msum, float* __restrict__ mcnt)
{
    int m = blockIdx.x * blockDim.x + threadIdx.x;
    if (m >= n_mol) return;
    int lo = 0, hi = n_atoms;
    while (lo < hi) { int mid = (lo + hi) >> 1; if (batch[mid] < m) lo = mid + 1; else hi = mid; }
    int start = lo;
    hi = n_atoms;
    while (lo < hi) { int mid = (lo + hi) >> 1; if (batch[mid] < m + 1) lo = mid + 1; else hi = mid; }
    int end = lo;
    float s = 0.f;
    for (int i = start; i < end; ++i) s += atom_out[i];
    msum[m] = s;
    mcnt[m] = (float)(end - start);
}

// ---------------------------------------------------------------------------
// Kernel 3: per-atom charge correction (in place on d_out)
// ---------------------------------------------------------------------------
__global__ void __launch_bounds__(256)
k_correct(float* __restrict__ out, const int* __restrict__ batch,
          const float* __restrict__ charge,
          const float* __restrict__ msum,
          const float* __restrict__ mcnt, int n_atoms)
{
    int a = blockIdx.x * blockDim.x + threadIdx.x;
    if (a >= n_atoms) return;
    int m = batch[a];
    out[a] = out[a] + (charge[m] - msum[m]) / mcnt[m];
}

// ---------------------------------------------------------------------------
extern "C" void kernel_launch(void* const* d_in, const int* in_sizes, int n_in,
                              void* d_out, int out_size, void* d_ws, size_t ws_size,
                              hipStream_t stream)
{
    const float* x      = (const float*)d_in[0];
    const int*   batch  = (const int*)d_in[1];
    const float* charge = (const float*)d_in[2];
    const float* W1     = (const float*)d_in[3];
    const float* b1     = (const float*)d_in[4];
    const float* W2     = (const float*)d_in[5];
    const float* b2     = (const float*)d_in[6];

    int n_atoms = in_sizes[1];
    int n_mol   = in_sizes[2];

    float* out  = (float*)d_out;
    float* msum = (float*)d_ws;
    float* mcnt = msum + n_mol;

    int nblk = (n_atoms + MT - 1) / MT;
    k_mlp<<<nblk, 256, 0, stream>>>(x, W1, b1, W2, b2, out, n_atoms);
    k_molsum<<<(n_mol + 255) / 256, 256, 0, stream>>>(out, batch, n_atoms, n_mol, msum, mcnt);
    k_correct<<<(n_atoms + 255) / 256, 256, 0, stream>>>(out, batch, charge, msum, mcnt, n_atoms);
}
